// EncLayer_11673721110549
// MI455X (gfx1250) — matrix operand order repaired and checked by reading, compiled, not run on hardware
//
#include <hip/hip_runtime.h>
#include <cstdint>

#define BB   4
#define NN   1024
#define KK   48
#define HH   128
#define EE   128
#define FFD  512
#define NIN  384            // H + E + H
#define EPSF 1e-5f
#define SCLF 30.0f

typedef __attribute__((ext_vector_type(16))) __bf16 v16bf;
typedef __attribute__((ext_vector_type(8)))  float  v8f;

union Frag16 { v16bf v; unsigned int u[8]; };

__device__ __forceinline__ uint16_t f2bf(float x) {
    unsigned int u = __float_as_uint(x);
    unsigned int r = (u + 0x7FFFu + ((u >> 16) & 1u)) >> 16;
    return (uint16_t)r;
}

__device__ __forceinline__ float gelu_exact(float x) {
    return 0.5f * x * (1.0f + erff(x * 0.70710678118654752f));
}

__device__ __forceinline__ v8f vzero() {
    v8f z = {0.f,0.f,0.f,0.f,0.f,0.f,0.f,0.f};
    return z;
}

__device__ __forceinline__ v8f wmma_bf16(v16bf a, v16bf b, v8f c) {
    // D = A(16x32 bf16) * B(32x16 bf16) + C(16x16 f32)
    return __builtin_amdgcn_wmma_f32_16x16x32_bf16(false, a, false, b, (short)0, c, false, false);
}

// A fragment 16x32 from row-major LDS tile [rows][ld] (bf16).
// Layout (ISA 7.12.2): lanes0-15 row M=lane, VGPR0-3 K=0..7, VGPR4-7 K=16..23;
// lanes16-31 same rows, K=8..15 / K=24..31.
__device__ __forceinline__ v16bf load_a_frag(const uint16_t* lds, int ld,
                                             int row0, int k0, int lane) {
    Frag16 f;
    int m = lane & 15, half = lane >> 4;
    const uint16_t* base = lds + (size_t)(row0 + m) * ld + k0 + half * 8;
#pragma unroll
    for (int j = 0; j < 4; ++j) f.u[j]     = *(const unsigned int*)(base + 2 * j);
#pragma unroll
    for (int j = 0; j < 4; ++j) f.u[4 + j] = *(const unsigned int*)(base + 16 + 2 * j);
    return f.v;
}

// B fragment 32x16 from bf16 weights stored [out][in] row-major (torch Linear).
// B[k][n] = W[n][k]: lane n = lane&15, lanes0-15 K=0..7 / K=16..23, lanes16-31 K=8..15 / K=24..31,
// 2 consecutive K per VGPR -> contiguous 32-bit loads.
__device__ __forceinline__ v16bf load_b_frag(const uint16_t* w, int Kin,
                                             int n0, int k0, int lane) {
    Frag16 f;
    int n = lane & 15, half = lane >> 4;
    const uint16_t* base = w + (size_t)(n0 + n) * Kin + k0 + half * 8;
#pragma unroll
    for (int j = 0; j < 4; ++j) f.u[j]     = *(const unsigned int*)(base + 2 * j);
#pragma unroll
    for (int j = 0; j < 4; ++j) f.u[4 + j] = *(const unsigned int*)(base + 16 + 2 * j);
    return f.v;
}

// ---------------- fp32 -> bf16 weight conversion ----------------
__global__ void cvt_bf16_kernel(const float* __restrict__ src,
                                uint16_t* __restrict__ dst, int n) {
    int i = blockIdx.x * blockDim.x + threadIdx.x;
    if (i < n) dst[i] = f2bf(src[i]);
}

// ---------------- row LayerNorm (H=128) -> bf16 ----------------
__global__ __launch_bounds__(256) void ln_rows_kernel(
    const float* __restrict__ x, const float* __restrict__ g,
    const float* __restrict__ b, uint16_t* __restrict__ out, int nrows) {
    int wave = threadIdx.x >> 5, lane = threadIdx.x & 31;
    int row = blockIdx.x * 8 + wave;
    if (row >= nrows) return;
    const float* xr = x + (size_t)row * HH;
    float v[4], s = 0.f, ss = 0.f;
#pragma unroll
    for (int j = 0; j < 4; ++j) { v[j] = xr[lane * 4 + j]; s += v[j]; ss += v[j] * v[j]; }
#pragma unroll
    for (int off = 16; off > 0; off >>= 1) {
        s  += __shfl_xor(s,  off, 32);
        ss += __shfl_xor(ss, off, 32);
    }
    float mu   = s * (1.0f / HH);
    float var  = ss * (1.0f / HH) - mu * mu;
    float rstd = rsqrtf(var + EPSF);
    uint16_t* orow = out + (size_t)row * HH;
#pragma unroll
    for (int j = 0; j < 4; ++j) {
        int c = lane * 4 + j;
        orow[c] = f2bf((v[j] - mu) * rstd * g[c] + b[c]);
    }
}

// ---------------- node message MLP + masked K-reduction ----------------
__global__ __launch_bounds__(256) void node_msg_kernel(
    const float* __restrict__ hV, const float* __restrict__ hE,
    const int* __restrict__ Eidx, const float* __restrict__ maskA,
    const uint16_t* __restrict__ hVn,
    const uint16_t* __restrict__ W1, const float* __restrict__ b1,
    const uint16_t* __restrict__ W2, const float* __restrict__ b2,
    const uint16_t* __restrict__ W3, const float* __restrict__ b3,
    const float* __restrict__ resw, float* __restrict__ hVmid)
{
    __shared__ uint16_t sA[KK * NIN];   // h_EV tile, 36.9 KB
    __shared__ uint16_t sX[KK * HH];
    __shared__ uint16_t sY[KK * HH];
    __shared__ float    sM[KK];

    int tid  = threadIdx.x;
    int node = blockIdx.x;              // b*N + n
    int bN   = (node / NN) * NN;

    // ---- pack: [ hVn(b,n) | hE(b,n,k,:) | hVn(b,idx) ] as bf16 ----
    const uint16_t* hvr = hVn + (size_t)node * HH;
    for (int i = tid; i < KK * HH; i += 256) {
        int k = i >> 7, c = i & 127;
        sA[k * NIN + c] = hvr[c];
    }
    const float* her = hE + (size_t)node * KK * EE;
    for (int i = tid; i < KK * EE; i += 256) {
        int k = i >> 7, c = i & 127;
        sA[k * NIN + HH + c] = f2bf(her[(size_t)k * EE + c]);
    }
    const int* idxr = Eidx + (size_t)node * KK;
    for (int i = tid; i < KK * HH; i += 256) {
        int k = i >> 7, c = i & 127;
        int nb = idxr[k];
        sA[k * NIN + 2 * HH + c] = hVn[(size_t)(bN + nb) * HH + c];
    }
    if (tid < KK) sM[tid] = maskA[(size_t)node * KK + tid];
    __syncthreads();

    int wave = tid >> 5, lane = tid & 31;
    int half = lane >> 4;
    int col  = wave * 16 + (lane & 15);

    // ---- GEMM1: [48x384] x [384x128], GELU ----
    v8f a0 = vzero(), a1 = vzero(), a2 = vzero();
    for (int kt = 0; kt < NIN; kt += 32) {
        v16bf bf = load_b_frag(W1, NIN, wave * 16, kt, lane);
        a0 = wmma_bf16(load_a_frag(sA, NIN,  0, kt, lane), bf, a0);
        a1 = wmma_bf16(load_a_frag(sA, NIN, 16, kt, lane), bf, a1);
        a2 = wmma_bf16(load_a_frag(sA, NIN, 32, kt, lane), bf, a2);
    }
    {
        float bias = b1[col];
#pragma unroll
        for (int r = 0; r < 8; ++r) {
            sX[( 0 + half * 8 + r) * HH + col] = f2bf(gelu_exact(a0[r] + bias));
            sX[(16 + half * 8 + r) * HH + col] = f2bf(gelu_exact(a1[r] + bias));
            sX[(32 + half * 8 + r) * HH + col] = f2bf(gelu_exact(a2[r] + bias));
        }
    }
    __syncthreads();

    // ---- GEMM2: [48x128] x [128x128], GELU ----
    a0 = vzero(); a1 = vzero(); a2 = vzero();
    for (int kt = 0; kt < HH; kt += 32) {
        v16bf bf = load_b_frag(W2, HH, wave * 16, kt, lane);
        a0 = wmma_bf16(load_a_frag(sX, HH,  0, kt, lane), bf, a0);
        a1 = wmma_bf16(load_a_frag(sX, HH, 16, kt, lane), bf, a1);
        a2 = wmma_bf16(load_a_frag(sX, HH, 32, kt, lane), bf, a2);
    }
    {
        float bias = b2[col];
#pragma unroll
        for (int r = 0; r < 8; ++r) {
            sY[( 0 + half * 8 + r) * HH + col] = f2bf(gelu_exact(a0[r] + bias));
            sY[(16 + half * 8 + r) * HH + col] = f2bf(gelu_exact(a1[r] + bias));
            sY[(32 + half * 8 + r) * HH + col] = f2bf(gelu_exact(a2[r] + bias));
        }
    }
    __syncthreads();

    // ---- GEMM3: [48x128] x [128x128], mask + sum over K ----
    a0 = vzero(); a1 = vzero(); a2 = vzero();
    for (int kt = 0; kt < HH; kt += 32) {
        v16bf bf = load_b_frag(W3, HH, wave * 16, kt, lane);
        a0 = wmma_bf16(load_a_frag(sY, HH,  0, kt, lane), bf, a0);
        a1 = wmma_bf16(load_a_frag(sY, HH, 16, kt, lane), bf, a1);
        a2 = wmma_bf16(load_a_frag(sY, HH, 32, kt, lane), bf, a2);
    }
    float bias = b3[col];
    float partial = 0.f;
#pragma unroll
    for (int r = 0; r < 8; ++r) {
        partial += sM[ 0 + half * 8 + r] * (a0[r] + bias);
        partial += sM[16 + half * 8 + r] * (a1[r] + bias);
        partial += sM[32 + half * 8 + r] * (a2[r] + bias);
    }
    partial += __shfl_xor(partial, 16, 32);   // combine M=0..7 with M=8..15 halves
    if (half == 0) {
        float rw  = resw[0];
        float out = hV[(size_t)node * HH + col] + (partial / SCLF) * rw;
        hVmid[(size_t)node * HH + col] = out;
    }
}

// ---------------- position-wise FFN (LN -> Win -> GELU -> Wout) ----------------
__global__ __launch_bounds__(256) void ffn_kernel(
    const float* __restrict__ hVmid, const float* __restrict__ maskV,
    const float* __restrict__ lng, const float* __restrict__ lnb,
    const uint16_t* __restrict__ Win,  const float* __restrict__ bin,
    const uint16_t* __restrict__ Wout, const float* __restrict__ bout,
    const float* __restrict__ resw,
    float* __restrict__ outV, uint16_t* __restrict__ hVf)
{
    __shared__ uint16_t sA[32 * HH];    // LN'd rows, bf16
    __shared__ uint16_t sX[32 * FFD];   // hidden, bf16

    int tid = threadIdx.x, wave = tid >> 5, lane = tid & 31;
    int half = lane >> 4;
    int row0 = blockIdx.x * 32;

    // LN: each wave does 4 rows, 4 cols per lane
#pragma unroll
    for (int rr = 0; rr < 4; ++rr) {
        int row = wave * 4 + rr;
        const float* xr = hVmid + (size_t)(row0 + row) * HH;
        float v[4], s = 0.f, ss = 0.f;
#pragma unroll
        for (int j = 0; j < 4; ++j) { v[j] = xr[lane * 4 + j]; s += v[j]; ss += v[j] * v[j]; }
#pragma unroll
        for (int off = 16; off > 0; off >>= 1) {
            s  += __shfl_xor(s,  off, 32);
            ss += __shfl_xor(ss, off, 32);
        }
        float mu   = s * (1.0f / HH);
        float var  = ss * (1.0f / HH) - mu * mu;
        float rstd = rsqrtf(var + EPSF);
#pragma unroll
        for (int j = 0; j < 4; ++j) {
            int c = lane * 4 + j;
            sA[row * HH + c] = f2bf((v[j] - mu) * rstd * lng[c] + lnb[c]);
        }
    }
    __syncthreads();

    // GEMM1: [32x128] x [128x512]; wave covers 4 col-tiles
    for (int s = 0; s < 4; ++s) {
        int n0 = (wave + 8 * s) * 16;
        v8f a0 = vzero(), a1 = vzero();
        for (int kt = 0; kt < HH; kt += 32) {
            v16bf bf = load_b_frag(Win, HH, n0, kt, lane);
            a0 = wmma_bf16(load_a_frag(sA, HH,  0, kt, lane), bf, a0);
            a1 = wmma_bf16(load_a_frag(sA, HH, 16, kt, lane), bf, a1);
        }
        int c = n0 + (lane & 15);
        float bias = bin[c];
#pragma unroll
        for (int r = 0; r < 8; ++r) {
            sX[( 0 + half * 8 + r) * FFD + c] = f2bf(gelu_exact(a0[r] + bias));
            sX[(16 + half * 8 + r) * FFD + c] = f2bf(gelu_exact(a1[r] + bias));
        }
    }
    __syncthreads();

    // GEMM2: [32x512] x [512x128]
    v8f acc[2]; acc[0] = vzero(); acc[1] = vzero();
    for (int kt = 0; kt < FFD; kt += 32) {
        v16bf bf = load_b_frag(Wout, FFD, wave * 16, kt, lane);
        acc[0] = wmma_bf16(load_a_frag(sX, FFD,  0, kt, lane), bf, acc[0]);
        acc[1] = wmma_bf16(load_a_frag(sX, FFD, 16, kt, lane), bf, acc[1]);
    }
    int col = wave * 16 + (lane & 15);
    float bias = bout[col];
    float rw = resw[0];
#pragma unroll
    for (int rt = 0; rt < 2; ++rt) {
#pragma unroll
        for (int r = 0; r < 8; ++r) {
            int gr = row0 + rt * 16 + half * 8 + r;
            float dh = acc[rt][r] + bias;
            float o  = (hVmid[(size_t)gr * HH + col] + rw * dh) * maskV[gr];
            outV[(size_t)gr * HH + col] = o;
            hVf[(size_t)gr * HH + col]  = f2bf(o);
        }
    }
}

// ---------------- edge message MLP + residual ----------------
__global__ __launch_bounds__(256) void edge_msg_kernel(
    const float* __restrict__ hE, const int* __restrict__ Eidx,
    const uint16_t* __restrict__ hVf,
    const float* __restrict__ lng, const float* __restrict__ lnb,
    const uint16_t* __restrict__ W11, const float* __restrict__ b11,
    const uint16_t* __restrict__ W12, const float* __restrict__ b12,
    const uint16_t* __restrict__ W13, const float* __restrict__ b13,
    float* __restrict__ outE)
{
    __shared__ uint16_t sA[KK * NIN];
    __shared__ uint16_t sX[KK * HH];
    __shared__ uint16_t sY[KK * HH];

    int tid  = threadIdx.x;
    int node = blockIdx.x;
    int bN   = (node / NN) * NN;
    int wave = tid >> 5, lane = tid & 31;
    int half = lane >> 4;

    // broadcast h_V(b,n) and gather h_V(b,idx)
    const uint16_t* hvr = hVf + (size_t)node * HH;
    for (int i = tid; i < KK * HH; i += 256) {
        int k = i >> 7, c = i & 127;
        sA[k * NIN + c] = hvr[c];
    }
    const int* idxr = Eidx + (size_t)node * KK;
    for (int i = tid; i < KK * HH; i += 256) {
        int k = i >> 7, c = i & 127;
        int nb = idxr[k];
        sA[k * NIN + 2 * HH + c] = hVf[(size_t)(bN + nb) * HH + c];
    }
    // LN(h_E row) into middle columns: wave handles rows wave, wave+8, ...
    for (int k = wave; k < KK; k += 8) {
        const float* xr = hE + ((size_t)node * KK + k) * EE;
        float v[4], s = 0.f, ss = 0.f;
#pragma unroll
        for (int j = 0; j < 4; ++j) { v[j] = xr[lane * 4 + j]; s += v[j]; ss += v[j] * v[j]; }
#pragma unroll
        for (int off = 16; off > 0; off >>= 1) {
            s  += __shfl_xor(s,  off, 32);
            ss += __shfl_xor(ss, off, 32);
        }
        float mu   = s * (1.0f / EE);
        float var  = ss * (1.0f / EE) - mu * mu;
        float rstd = rsqrtf(var + EPSF);
#pragma unroll
        for (int j = 0; j < 4; ++j) {
            int c = lane * 4 + j;
            sA[k * NIN + HH + c] = f2bf((v[j] - mu) * rstd * lng[c] + lnb[c]);
        }
    }
    __syncthreads();

    int col = wave * 16 + (lane & 15);

    v8f a0 = vzero(), a1 = vzero(), a2 = vzero();
    for (int kt = 0; kt < NIN; kt += 32) {
        v16bf bf = load_b_frag(W11, NIN, wave * 16, kt, lane);
        a0 = wmma_bf16(load_a_frag(sA, NIN,  0, kt, lane), bf, a0);
        a1 = wmma_bf16(load_a_frag(sA, NIN, 16, kt, lane), bf, a1);
        a2 = wmma_bf16(load_a_frag(sA, NIN, 32, kt, lane), bf, a2);
    }
    {
        float bias = b11[col];
#pragma unroll
        for (int r = 0; r < 8; ++r) {
            sX[( 0 + half * 8 + r) * HH + col] = f2bf(gelu_exact(a0[r] + bias));
            sX[(16 + half * 8 + r) * HH + col] = f2bf(gelu_exact(a1[r] + bias));
            sX[(32 + half * 8 + r) * HH + col] = f2bf(gelu_exact(a2[r] + bias));
        }
    }
    __syncthreads();

    a0 = vzero(); a1 = vzero(); a2 = vzero();
    for (int kt = 0; kt < HH; kt += 32) {
        v16bf bf = load_b_frag(W12, HH, wave * 16, kt, lane);
        a0 = wmma_bf16(load_a_frag(sX, HH,  0, kt, lane), bf, a0);
        a1 = wmma_bf16(load_a_frag(sX, HH, 16, kt, lane), bf, a1);
        a2 = wmma_bf16(load_a_frag(sX, HH, 32, kt, lane), bf, a2);
    }
    {
        float bias = b12[col];
#pragma unroll
        for (int r = 0; r < 8; ++r) {
            sY[( 0 + half * 8 + r) * HH + col] = f2bf(gelu_exact(a0[r] + bias));
            sY[(16 + half * 8 + r) * HH + col] = f2bf(gelu_exact(a1[r] + bias));
            sY[(32 + half * 8 + r) * HH + col] = f2bf(gelu_exact(a2[r] + bias));
        }
    }
    __syncthreads();

    a0 = vzero(); a1 = vzero(); a2 = vzero();
    for (int kt = 0; kt < HH; kt += 32) {
        v16bf bf = load_b_frag(W13, HH, wave * 16, kt, lane);
        a0 = wmma_bf16(load_a_frag(sY, HH,  0, kt, lane), bf, a0);
        a1 = wmma_bf16(load_a_frag(sY, HH, 16, kt, lane), bf, a1);
        a2 = wmma_bf16(load_a_frag(sY, HH, 32, kt, lane), bf, a2);
    }
    float bias = b13[col];
    const float* hres = hE + (size_t)node * KK * EE;
    float* oe = outE + (size_t)node * KK * HH;
#pragma unroll
    for (int r = 0; r < 8; ++r) {
        int r0 =  0 + half * 8 + r;
        int r1 = 16 + half * 8 + r;
        int r2 = 32 + half * 8 + r;
        oe[(size_t)r0 * HH + col] = hres[(size_t)r0 * EE + col] + (a0[r] + bias);
        oe[(size_t)r1 * HH + col] = hres[(size_t)r1 * EE + col] + (a1[r] + bias);
        oe[(size_t)r2 * HH + col] = hres[(size_t)r2 * EE + col] + (a2[r] + bias);
    }
}

extern "C" void kernel_launch(void* const* d_in, const int* in_sizes, int n_in,
                              void* d_out, int out_size, void* d_ws, size_t ws_size,
                              hipStream_t stream) {
    (void)in_sizes; (void)n_in; (void)out_size; (void)ws_size;
    const float* hV    = (const float*)d_in[0];
    const float* hE    = (const float*)d_in[1];
    const int*   Eidx  = (const int*)d_in[2];
    const float* maskV = (const float*)d_in[3];
    const float* maskA = (const float*)d_in[4];
    const float* W1w   = (const float*)d_in[5];  const float* W1b  = (const float*)d_in[6];
    const float* W2w   = (const float*)d_in[7];  const float* W2b  = (const float*)d_in[8];
    const float* W3w   = (const float*)d_in[9];  const float* W3b  = (const float*)d_in[10];
    const float* W11w  = (const float*)d_in[11]; const float* W11b = (const float*)d_in[12];
    const float* W12w  = (const float*)d_in[13]; const float* W12b = (const float*)d_in[14];
    const float* W13w  = (const float*)d_in[15]; const float* W13b = (const float*)d_in[16];
    const float* ln0g  = (const float*)d_in[17]; const float* ln0b = (const float*)d_in[18];
    const float* ln1g  = (const float*)d_in[19]; const float* ln1b = (const float*)d_in[20];
    const float* ln2g  = (const float*)d_in[21]; const float* ln2b = (const float*)d_in[22];
    const float* Winw  = (const float*)d_in[23]; const float* Winb = (const float*)d_in[24];
    const float* Woutw = (const float*)d_in[25]; const float* Woutb= (const float*)d_in[26];
    const float* resw  = (const float*)d_in[27];

    // workspace layout (bf16 weights + bf16 activations + fp32 h_V_mid) ~4.7 MB
    uint16_t* W1bf   = (uint16_t*)d_ws;
    uint16_t* W2bf   = W1bf  + 128 * 384;
    uint16_t* W3bf   = W2bf  + 128 * 128;
    uint16_t* W11bf  = W3bf  + 128 * 128;
    uint16_t* W12bf  = W11bf + 128 * 384;
    uint16_t* W13bf  = W12bf + 128 * 128;
    uint16_t* Winbf  = W13bf + 128 * 128;
    uint16_t* Woutbf = Winbf + 512 * 128;
    uint16_t* hVnbf  = Woutbf + 128 * 512;
    uint16_t* hVfbf  = hVnbf + (size_t)BB * NN * HH;
    float*    hVmid  = (float*)(hVfbf + (size_t)BB * NN * HH);

    float* outV = (float*)d_out;
    float* outE = outV + (size_t)BB * NN * HH;

    auto cvt = [&](const float* s, uint16_t* d, int n) {
        cvt_bf16_kernel<<<(n + 255) / 256, 256, 0, stream>>>(s, d, n);
    };
    cvt(W1w,  W1bf,  128 * 384);
    cvt(W2w,  W2bf,  128 * 128);
    cvt(W3w,  W3bf,  128 * 128);
    cvt(W11w, W11bf, 128 * 384);
    cvt(W12w, W12bf, 128 * 128);
    cvt(W13w, W13bf, 128 * 128);
    cvt(Winw, Winbf, 512 * 128);
    cvt(Woutw,Woutbf,128 * 512);

    ln_rows_kernel<<<(BB * NN) / 8, 256, 0, stream>>>(hV, ln0g, ln0b, hVnbf, BB * NN);

    node_msg_kernel<<<BB * NN, 256, 0, stream>>>(
        hV, hE, Eidx, maskA, hVnbf, W1bf, W1b, W2bf, W2b, W3bf, W3b, resw, hVmid);

    ffn_kernel<<<(BB * NN) / 32, 256, 0, stream>>>(
        hVmid, maskV, ln1g, ln1b, Winbf, Winb, Woutbf, Woutb, resw, outV, hVfbf);

    edge_msg_kernel<<<BB * NN, 256, 0, stream>>>(
        hE, Eidx, hVfbf, ln2g, ln2b, W11bf, W11b, W12bf, W12b, W13bf, W13b, outE);
}
